// GraphSAGE_81870666596807
// MI455X (gfx1250) — matrix operand order repaired and checked by reading, not compiled, not run on hardware
//
#include <hip/hip_runtime.h>
#include <hip/hip_bf16.h>

static constexpr int kD = 128;   // feature width (D_IN == D_H == 128)

typedef __bf16 bf16_t;
typedef bf16_t v16bf __attribute__((ext_vector_type(16)));
typedef bf16_t v2bf  __attribute__((ext_vector_type(2)));
typedef float  v8f   __attribute__((ext_vector_type(8)));

typedef union {
    v16bf        bf;
    unsigned int u[8];
    uint4        q[2];
} Frag16;

// f32x2 -> packed bf16 via native fptrunc (backend selects hw cvt when available)
__device__ __forceinline__ unsigned int pk_bf16(float lo, float hi) {
    union { v2bf b; unsigned int u; } c;
    c.b[0] = (bf16_t)lo;
    c.b[1] = (bf16_t)hi;
    return c.u;
}

__device__ __forceinline__ void wait_asynccnt0() {
#if defined(__has_builtin)
#if __has_builtin(__builtin_amdgcn_s_wait_asynccnt)
    __builtin_amdgcn_s_wait_asynccnt(0);
    return;
#else
    asm volatile("s_wait_asynccnt 0x0" ::: "memory");
    return;
#endif
#else
    asm volatile("s_wait_asynccnt 0x0" ::: "memory");
#endif
}

// ---------------- utility kernels ----------------

__global__ void k_zero(float* p, long long n) {
    long long i = (long long)blockIdx.x * blockDim.x + threadIdx.x;
    long long stride = (long long)gridDim.x * blockDim.x;
    for (; i < n; i += stride) p[i] = 0.0f;
}

__global__ void k_deg(const int* __restrict__ dst, float* __restrict__ deg, int E) {
    int i = blockIdx.x * blockDim.x + threadIdx.x;
    if (i < E) atomicAdd(&deg[dst[i]], 1.0f);
}

__global__ void k_invdeg(const float* __restrict__ deg, float* __restrict__ inv, int n) {
    int i = blockIdx.x * blockDim.x + threadIdx.x;
    if (i < n) inv[i] = 1.0f / fmaxf(deg[i], 1.0f);
}

// fp32 -> bf16 copy (used for x and the four 128x128 weight matrices)
__global__ void k_cvt_bf16(const float* __restrict__ src, bf16_t* __restrict__ dst,
                           long long n) {
    long long i = (long long)blockIdx.x * blockDim.x + threadIdx.x;
    long long stride = (long long)gridDim.x * blockDim.x;
    for (; i < n; i += stride) dst[i] = (bf16_t)src[i];
}

// ---------------- edge aggregation: agg[dst] += h_bf16[src] * inv_deg[dst] ----------------
// One wave per edge; each lane moves 4 bf16 features (8B read) and does 4 f32 atomics.
// Features (bf16) and agg (f32) are both L2-resident (<< 192 MB).
__global__ void k_aggregate(const unsigned short* __restrict__ hb, const int* __restrict__ src,
                            const int* __restrict__ dst, const float* __restrict__ inv_deg,
                            float* __restrict__ agg, int E) {
    int t = blockIdx.x * blockDim.x + threadIdx.x;
    int e = t >> 5;
    int lane = t & 31;
    if (e >= E) return;
    int s = src[e], d = dst[e];
    float w = inv_deg[d];
    int f = lane * 4;
    uint2 p = *(const uint2*)(hb + (long long)s * kD + f);
    float v0 = __uint_as_float(p.x << 16);
    float v1 = __uint_as_float(p.x & 0xFFFF0000u);
    float v2 = __uint_as_float(p.y << 16);
    float v3 = __uint_as_float(p.y & 0xFFFF0000u);
    float* base = agg + (long long)d * kD + f;
    atomicAdd(base + 0, v0 * w);
    atomicAdd(base + 1, v1 * w);
    atomicAdd(base + 2, v2 * w);
    atomicAdd(base + 3, v3 * w);
}

// ---------------- fused SAGE layer: agg @ Wl^T + b + h @ Wr^T, relu, bf16 out ----------------
// Block = 128 threads (4 waves). w_bf = [Wl | Wr] contiguous bf16; staged into one
// 64 KB LDS region with GLOBAL_LOAD_ASYNC_TO_LDS_B128 (ASYNCcnt, no VGPR round-trip).
// Each wave produces one 16x128 output tile via 64 v_wmma_f32_16x16x32_bf16.
// agg already carries the 1/deg scaling (folded into the scatter).
__global__ void __launch_bounds__(128, 1)
k_sage_wmma(const float* __restrict__ agg, const unsigned short* __restrict__ hb,
            const unsigned short* __restrict__ w_bf,   // [128x128 Wl][128x128 Wr] bf16
            const float* __restrict__ bias, bf16_t* __restrict__ outb, int nNodes) {
    __shared__ __align__(16) unsigned short lds_w[2 * kD * kD];   // 64 KB

    int tid = threadIdx.x;
    {   // async global->LDS weight stage: 32 x B128 per lane, tracked on ASYNCcnt
        unsigned ldsBase = (unsigned)(unsigned long long)(&lds_w[0]); // LDS byte offset
        #pragma unroll
        for (int i = 0; i < 32; ++i) {
            unsigned byteOff = (unsigned)(tid * 16 + i * 2048);
            unsigned ldsAddr = ldsBase + byteOff;
            asm volatile("global_load_async_to_lds_b128 %0, %1, %2"
                         :: "v"(ldsAddr), "v"(byteOff), "s"(w_bf)
                         : "memory");
        }
        wait_asynccnt0();
    }
    __syncthreads();

    const unsigned short* lds_wl = lds_w;
    const unsigned short* lds_wr = lds_w + kD * kD;

    int wave = tid >> 5;
    int lane = tid & 31;
    int mTile = blockIdx.x * 4 + wave;           // wave-uniform
    if (mTile * 16 >= nNodes) return;            // whole wave exits -> EXEC stays all-1s

    int hv  = lane >> 4;                         // lane half: 0 or 1
    int l16 = lane & 15;
    int mBase = mTile * 16;
    int rowA  = mBase + l16;                     // A-matrix row owned by this lane

    const float*          aggRow = agg + (long long)rowA * kD;
    const unsigned short* hRow   = hb  + (long long)rowA * kD;

    v8f acc[8] = {};                             // 8 N-tiles of 16 -> full 128-wide output row

    #pragma unroll
    for (int ks = 0; ks < 4; ++ks) {             // K = 128 in steps of 32
        int kbA = ks * 32 + hv * 8;
        // A fragment, aggregated path (f32 -> bf16 pack):
        // element i -> k = (i&7) + 8*half + 16*(i>>3) + 32*ks
        float4 a0 = *(const float4*)(aggRow + kbA);
        float4 a1 = *(const float4*)(aggRow + kbA + 4);
        float4 a2 = *(const float4*)(aggRow + kbA + 16);
        float4 a3 = *(const float4*)(aggRow + kbA + 20);
        Frag16 fa;
        fa.u[0] = pk_bf16(a0.x, a0.y);
        fa.u[1] = pk_bf16(a0.z, a0.w);
        fa.u[2] = pk_bf16(a1.x, a1.y);
        fa.u[3] = pk_bf16(a1.z, a1.w);
        fa.u[4] = pk_bf16(a2.x, a2.y);
        fa.u[5] = pk_bf16(a2.z, a2.w);
        fa.u[6] = pk_bf16(a3.x, a3.y);
        fa.u[7] = pk_bf16(a3.z, a3.w);
        // A fragment, root path: h already bf16 in memory in exactly the fragment
        // pair order -> two raw 16B loads, zero conversion ALU.
        Frag16 fh;
        fh.q[0] = *(const uint4*)(hRow + kbA);
        fh.q[1] = *(const uint4*)(hRow + kbA + 16);

        int kbB = ks * 32 + hv * 8;              // B element i -> k = (i&7) + 8*half + 16*(i>>3) + 32*ks
        #pragma unroll
        for (int nt = 0; nt < 8; ++nt) {
            int nRow = nt * 16 + l16;            // B column owned by this lane
            Frag16 bl, br;
            const uint4* pl = (const uint4*)(lds_wl + nRow * kD + kbB);
            bl.q[0] = pl[0]; bl.q[1] = pl[2];
            const uint4* pr = (const uint4*)(lds_wr + nRow * kD + kbB);
            br.q[0] = pr[0]; br.q[1] = pr[2];
            acc[nt] = __builtin_amdgcn_wmma_f32_16x16x32_bf16(
                false, fa.bf, false, bl.bf, (short)0, acc[nt], false, false);
            acc[nt] = __builtin_amdgcn_wmma_f32_16x16x32_bf16(
                false, fh.bf, false, br.bf, (short)0, acc[nt], false, false);
        }
    }

    // epilogue: + bias, relu, store bf16 (D layout: VGPR r -> row r + 8*half, col l16)
    #pragma unroll
    for (int nt = 0; nt < 8; ++nt) {
        int n = nt * 16 + l16;
        float bv = bias[n];
        #pragma unroll
        for (int r = 0; r < 8; ++r) {
            float v = fmaxf(acc[nt][r] + bv, 0.0f);
            outb[(long long)(mBase + r + hv * 8) * kD + n] = (bf16_t)v;
        }
    }
}

// ---------------- final layer (D_OUT=2) + log_softmax, fp32 ----------------
// One wave per node; each lane handles 4 features; shfl_xor reduction (wave32).
// agg already holds the neighbor mean (inv_deg folded into the scatter).
__global__ void k_final(const float* __restrict__ agg, const unsigned short* __restrict__ hb,
                        const float* __restrict__ wl2, const float* __restrict__ b2,
                        const float* __restrict__ wr2, float* __restrict__ out, int nNodes) {
    int t = blockIdx.x * blockDim.x + threadIdx.x;
    int node = t >> 5;
    int lane = t & 31;
    if (node >= nNodes) return;
    int f = lane * 4;
    float4 av = *(const float4*)(agg + (long long)node * kD + f);
    uint2 hp = *(const uint2*)(hb + (long long)node * kD + f);
    float h0 = __uint_as_float(hp.x << 16);
    float h1 = __uint_as_float(hp.x & 0xFFFF0000u);
    float h2 = __uint_as_float(hp.y << 16);
    float h3 = __uint_as_float(hp.y & 0xFFFF0000u);
    float4 l0 = *(const float4*)(wl2 + f);
    float4 l1 = *(const float4*)(wl2 + kD + f);
    float4 r0 = *(const float4*)(wr2 + f);
    float4 r1 = *(const float4*)(wr2 + kD + f);
    float o0 = av.x * l0.x + av.y * l0.y + av.z * l0.z + av.w * l0.w
             + h0 * r0.x + h1 * r0.y + h2 * r0.z + h3 * r0.w;
    float o1 = av.x * l1.x + av.y * l1.y + av.z * l1.z + av.w * l1.w
             + h0 * r1.x + h1 * r1.y + h2 * r1.z + h3 * r1.w;
    #pragma unroll
    for (int off = 16; off >= 1; off >>= 1) {
        o0 += __shfl_xor(o0, off, 32);
        o1 += __shfl_xor(o1, off, 32);
    }
    if (lane == 0) {
        o0 += b2[0]; o1 += b2[1];
        float m = fmaxf(o0, o1);
        float lse = m + logf(expf(o0 - m) + expf(o1 - m));
        out[(long long)node * 2 + 0] = o0 - lse;
        out[(long long)node * 2 + 1] = o1 - lse;
    }
}

// ---------------- host launcher ----------------

extern "C" void kernel_launch(void* const* d_in, const int* in_sizes, int n_in,
                              void* d_out, int out_size, void* d_ws, size_t ws_size,
                              hipStream_t stream) {
    const float* x    = (const float*)d_in[0];
    const float* w_l0 = (const float*)d_in[1];
    const float* b0   = (const float*)d_in[2];
    const float* w_r0 = (const float*)d_in[3];
    const float* w_l1 = (const float*)d_in[4];
    const float* b1   = (const float*)d_in[5];
    const float* w_r1 = (const float*)d_in[6];
    const float* w_l2 = (const float*)d_in[7];
    const float* b2   = (const float*)d_in[8];
    const float* w_r2 = (const float*)d_in[9];
    const int* esrc   = (const int*)d_in[10];
    const int* edst   = (const int*)d_in[11];
    float* out = (float*)d_out;

    const int N = in_sizes[0] / kD;   // 100000
    const int E = in_sizes[10];       // 1600000

    // workspace layout
    float* ws  = (float*)d_ws;
    float* deg = ws;                                   // N f32
    float* inv = deg + N;                              // N f32
    float* agg = inv + N;                              // N*128 f32 (L2-resident)
    unsigned short* xb  = (unsigned short*)(agg + (size_t)N * kD); // N*128 bf16
    unsigned short* h1b = xb + (size_t)N * kD;                     // N*128 bf16
    unsigned short* h2b = h1b + (size_t)N * kD;                    // N*128 bf16
    unsigned short* wbf = h2b + (size_t)N * kD;                    // [wl0|wr0|wl1|wr1] bf16
    unsigned short* w0b = wbf;                 // layer-0 [Wl|Wr], contiguous
    unsigned short* w1b = wbf + 2 * kD * kD;   // layer-1 [Wl|Wr], contiguous

    dim3 b256(256);
    const int aggBlocks  = (int)(((long long)E * 32 + 255) / 256);
    const int nTiles     = (N + 15) / 16;
    const int sageBlocks = (nTiles + 3) / 4;

    // degree / inv_deg (shared by all layers) + bf16 conversions
    k_zero<<<1024, b256, 0, stream>>>(deg, N);
    k_deg<<<(E + 255) / 256, b256, 0, stream>>>(edst, deg, E);
    k_invdeg<<<(N + 255) / 256, b256, 0, stream>>>(deg, inv, N);
    k_cvt_bf16<<<2048, b256, 0, stream>>>(x, (bf16_t*)xb, (long long)N * kD);
    k_cvt_bf16<<<64,   b256, 0, stream>>>(w_l0, (bf16_t*)w0b, kD * kD);
    k_cvt_bf16<<<64,   b256, 0, stream>>>(w_r0, (bf16_t*)(w0b + kD * kD), kD * kD);
    k_cvt_bf16<<<64,   b256, 0, stream>>>(w_l1, (bf16_t*)w1b, kD * kD);
    k_cvt_bf16<<<64,   b256, 0, stream>>>(w_r1, (bf16_t*)(w1b + kD * kD), kD * kD);

    // layer 0
    k_zero<<<2048, b256, 0, stream>>>(agg, (long long)N * kD);
    k_aggregate<<<aggBlocks, b256, 0, stream>>>(xb, esrc, edst, inv, agg, E);
    k_sage_wmma<<<sageBlocks, 128, 0, stream>>>(agg, xb, w0b, b0, (bf16_t*)h1b, N);

    // layer 1
    k_zero<<<2048, b256, 0, stream>>>(agg, (long long)N * kD);
    k_aggregate<<<aggBlocks, b256, 0, stream>>>(h1b, esrc, edst, inv, agg, E);
    k_sage_wmma<<<sageBlocks, 128, 0, stream>>>(agg, h1b, w1b, b1, (bf16_t*)h2b, N);

    // layer 2 (D_OUT = 2) + log_softmax, fp32
    k_zero<<<2048, b256, 0, stream>>>(agg, (long long)N * kD);
    k_aggregate<<<aggBlocks, b256, 0, stream>>>(h2b, esrc, edst, inv, agg, E);
    k_final<<<(int)(((long long)N * 32 + 255) / 256), b256, 0, stream>>>(
        agg, h2b, w_l2, b2, w_r2, out, N);
}
